// DeconBlock_71605694759688
// MI455X (gfx1250) — matrix operand order repaired and checked by reading, not compiled, not run on hardware
//
#include <hip/hip_runtime.h>
#include <hip/hip_bf16.h>
#include <math.h>

// ---------------------------------------------------------------------------
// DeconBlock forward for MI455X (gfx1250, wave32, WMMA).
// Single f16 im2col GEMM kernel for every conv (v_wmma_f32_16x16x32_f16):
//   - whole 64xK f16 weight panel staged ONCE in LDS (<=144KB of the WGP's
//     320KB), k-loop runs barrier-free: ds_load_b128 + global_load_b128 + wmma
//   - each wave owns a 32(M) x 64(N) tile -> 8 WMMA per 32-deep k-step,
//     B fragments reused across both M sub-tiles, f32 accumulate
// DCNv2 bilinear sampling emits the same [M,K] f16 layout, so its dense
// contraction is the identical GEMM.
// ---------------------------------------------------------------------------

typedef __attribute__((ext_vector_type(16))) _Float16 v16h;
typedef __attribute__((ext_vector_type(8)))  _Float16 v8h;
typedef __attribute__((ext_vector_type(8)))  float    v8f;

union AU { v16h v; _Float16 h[16]; };
union CU { v8f  v; float    f[8];  };

#define MB     4
#define WO     96
#define HWO    9216            // 96*96
#define MTOT   36864           // 4*9216, multiple of 128

// ------------------------- f32 -> f16 weight cast ---------------------------
__global__ __launch_bounds__(256)
void wcvt_f16(const float* __restrict__ src, _Float16* __restrict__ dst, int n)
{
    int i = blockIdx.x * blockDim.x + threadIdx.x;
    if (i < n) dst[i] = (_Float16)src[i];
}

// ------------------------------ im2col to f16 -------------------------------
// One block per output pixel m; threads stride over K. dst[m*K + k],
// k = ci*KH*KW + ky*KW + kx  (matches OIHW weight flattening).
template <int KH, int KW>
__global__ __launch_bounds__(256)
void im2col_f16(const float* __restrict__ src, _Float16* __restrict__ dst,
                int Cin, int Hin, int Win, int stride, int pad)
{
    constexpr int KK = KH * KW;
    const int K = Cin * KK;
    const int m = blockIdx.x;              // 0..MTOT-1
    const int b = m / HWO;
    const int p = m - b * HWO;
    const int oy = p / WO;
    const int ox = p - oy * WO;
    _Float16* drow = dst + (size_t)m * K;
    for (int k = threadIdx.x; k < K; k += 256) {
        const int ci = k / KK;
        const int r  = k - ci * KK;
        const int ky = r / KW;
        const int kx = r - ky * KW;
        const int iy = oy * stride - pad + ky;
        const int ix = ox * stride - pad + kx;
        float v = 0.0f;
        if (iy >= 0 && iy < Hin && ix >= 0 && ix < Win)
            v = src[((size_t)(b * Cin + ci) * Hin + iy) * Win + ix];
        drow[k] = (_Float16)v;
    }
}

// ------------------------------- WMMA GEMM ----------------------------------
// Block = 128 threads = 4 waves. Wave w owns M rows [(bx*4+w)*32, +32),
// block owns N cols [by*64, +64). Whole 64xK B panel staged once in LDS.
template <int K>
__global__ __launch_bounds__(128)
void gemm_wmma_f16(const _Float16* __restrict__ A,   // [MTOT, K] f16 im2col
                   const _Float16* __restrict__ Bw,  // [Cout, K] f16 weights
                   const float* __restrict__ bnp,    // [4,Cout] or null
                   const float* __restrict__ bias,   // [Cout] or null
                   const float* __restrict__ res,    // residual or null
                   float* __restrict__ dst,
                   int Cout, int relu)
{
    __shared__ __align__(32) _Float16 ldsB[64 * K];  // <= 144 KB of 320 KB/WGP

    const int lane  = threadIdx.x & 31;
    const int wv    = threadIdx.x >> 5;
    const int half  = lane >> 4;          // 0: lanes 0-15, 1: lanes 16-31
    const int laneN = lane & 15;

    const int mbase  = (blockIdx.x * 4 + wv) * 32;
    const int nblock = blockIdx.y * 64;

    // ---- stage full 64 x K weight panel into LDS (row-clamped, no branch;
    //      clamped rows only feed columns whose stores are guarded) ----
    {
        const int sn    = threadIdx.x >> 1;          // 0..63
        const int shalf = threadIdx.x & 1;           // which 8-wide phase
        int gn = nblock + sn;
        if (gn > Cout - 1) gn = Cout - 1;
        const _Float16* wp = Bw + (size_t)gn * K;
        _Float16* lp = &ldsB[sn * K];
        for (int k = shalf * 8; k < K; k += 16)
            *(v8h*)&lp[k] = *(const v8h*)&wp[k];
    }
    __syncthreads();

    const _Float16* ap0 = A + (size_t)(mbase + laneN) * K + (half << 3);
    const _Float16* ap1 = ap0 + (size_t)16 * K;

    CU c[2][4];
#pragma unroll
    for (int mt = 0; mt < 2; ++mt)
#pragma unroll
        for (int t = 0; t < 4; ++t)
#pragma unroll
            for (int i = 0; i < 8; ++i) c[mt][t].f[i] = 0.0f;

    for (int k0 = 0; k0 < K; k0 += 32) {
        // A fragments: lane<16 holds K {0..7,16..23}, lane>=16 {8..15,24..31}
        AU a0, a1;
        *(v8h*)&a0.h[0] = *(const v8h*)(ap0 + k0);
        *(v8h*)&a0.h[8] = *(const v8h*)(ap0 + k0 + 16);
        *(v8h*)&a1.h[0] = *(const v8h*)(ap1 + k0);
        *(v8h*)&a1.h[8] = *(const v8h*)(ap1 + k0 + 16);
#pragma unroll
        for (int t = 0; t < 4; ++t) {
            // B fragment: col n = t*16+laneN, K {0..7,16..23}(lanes<16)/{8..15,24..31}
            AU bm;
            const _Float16* bp = &ldsB[(t * 16 + laneN) * K + k0 + (half << 3)];
            *(v8h*)&bm.h[0] = *(const v8h*)bp;
            *(v8h*)&bm.h[8] = *(const v8h*)(bp + 16);
            c[0][t].v = __builtin_amdgcn_wmma_f32_16x16x32_f16(
                            false, a0.v, false, bm.v, (short)0, c[0][t].v, false, false);
            c[1][t].v = __builtin_amdgcn_wmma_f32_16x16x32_f16(
                            false, a1.v, false, bm.v, (short)0, c[1][t].v, false, false);
        }
    }

    // fused epilogue: (+bias) -> BN -> (+residual) -> LeakyReLU
#pragma unroll
    for (int t = 0; t < 4; ++t) {
        const int nW = nblock + t * 16 + laneN;
        if (nW >= Cout) continue;
        float inv = 1.0f, sh2 = 0.0f, bs = 0.0f;
        if (bnp) {
            const float g  = bnp[nW];
            const float be = bnp[Cout + nW];
            const float mn = bnp[2 * Cout + nW];
            const float vr = bnp[3 * Cout + nW];
            inv = g * rsqrtf(vr + 1e-5f);
            sh2 = be - mn * inv;
        }
        if (bias) bs = bias[nW];
#pragma unroll
        for (int mt = 0; mt < 2; ++mt) {
#pragma unroll
            for (int r = 0; r < 8; ++r) {
                const int mrow = mbase + mt * 16 + r + (half << 3);
                const int b    = mrow / HWO;
                const int rem  = mrow - b * HWO;
                const int oy   = rem / WO;
                const int ox   = rem - oy * WO;
                float val = c[mt][t].f[r] + bs;
                if (bnp) val = val * inv + sh2;
                const size_t oidx = ((size_t)(b * Cout + nW) * WO + oy) * WO + ox;
                if (res)  val += res[oidx];
                if (relu) val = (val >= 0.0f) ? val : 0.1f * val;
                dst[oidx] = val;
            }
        }
    }
}

// ----------------------- DCNv2 modulated bilinear sampling ------------------
// Writes the f16 im2col layout: A[m*1152 + (g*64+cg)*9 + k2]
__global__ __launch_bounds__(256)
void mdcn_sample(const float* __restrict__ h2, const float* __restrict__ off,
                 _Float16* __restrict__ valbuf)
{
    const int G = 2, K2 = 9, Cg = 64, Cm = 128, Kdim = 1152;
    const int total = MB * G * K2 * HWO;
    int idx = blockIdx.x * blockDim.x + threadIdx.x;
    if (idx >= total) return;

    const int p  = idx % HWO; int t = idx / HWO;
    const int k2 = t % K2;    t /= K2;
    const int g  = t % G;
    const int b  = t / G;
    const int oy = p / WO, ox = p - oy * WO;
    const int ch = g * K2 + k2;

    const size_t ob = (size_t)b * 54 * HWO;
    const float dy = off[ob + (size_t)ch        * HWO + p];
    const float dx = off[ob + (size_t)(18 + ch) * HWO + p];
    const float ml = off[ob + (size_t)(36 + ch) * HWO + p];
    const float mask = 1.0f / (1.0f + expf(-ml));

    const int ky = k2 / 3, kx = k2 - ky * 3;
    const float py = (float)(oy - 1 + ky) + dy;   // pad=1, stride=1
    const float px = (float)(ox - 1 + kx) + dx;
    const float y0f = floorf(py), x0f = floorf(px);
    const float wy = py - y0f, wx = px - x0f;
    const int y0 = (int)y0f, x0 = (int)x0f;
    const int y1 = y0 + 1,   x1 = x0 + 1;
    const bool vy0 = (y0 >= 0) && (y0 <= WO - 1), vy1 = (y1 >= 0) && (y1 <= WO - 1);
    const bool vx0 = (x0 >= 0) && (x0 <= WO - 1), vx1 = (x1 >= 0) && (x1 <= WO - 1);
    const int y0c = min(max(y0, 0), WO - 1), y1c = min(max(y1, 0), WO - 1);
    const int x0c = min(max(x0, 0), WO - 1), x1c = min(max(x1, 0), WO - 1);
    const float w00 = (1.0f - wy) * (1.0f - wx) * ((vy0 && vx0) ? 1.0f : 0.0f);
    const float w01 = (1.0f - wy) * wx          * ((vy0 && vx1) ? 1.0f : 0.0f);
    const float w10 = wy * (1.0f - wx)          * ((vy1 && vx0) ? 1.0f : 0.0f);
    const float w11 = wy * wx                   * ((vy1 && vx1) ? 1.0f : 0.0f);
    const int i00 = y0c * WO + x0c, i01 = y0c * WO + x1c;
    const int i10 = y1c * WO + x0c, i11 = y1c * WO + x1c;

    const size_t srcb = ((size_t)b * Cm + g * Cg) * HWO;
    const size_t dstb = ((size_t)b * HWO + p) * Kdim + (size_t)(g * Cg) * 9 + k2;
    for (int cg = 0; cg < Cg; ++cg) {
        const float* sp = h2 + srcb + (size_t)cg * HWO;
        const float v = w00 * sp[i00] + w01 * sp[i01] + w10 * sp[i10] + w11 * sp[i11];
        valbuf[dstb + (size_t)cg * 9] = (_Float16)(mask * v);
    }
}

// ------------------------------- launcher -----------------------------------
extern "C" void kernel_launch(void* const* d_in, const int* in_sizes, int n_in,
                              void* d_out, int out_size, void* d_ws, size_t ws_size,
                              hipStream_t stream) {
    (void)in_sizes; (void)n_in; (void)out_size; (void)ws_size;

    const float* x     = (const float*)d_in[0];
    const float* w_r1  = (const float*)d_in[1];
    const float* bn_r1 = (const float*)d_in[2];
    const float* w_r2  = (const float*)d_in[3];
    const float* bn_r2 = (const float*)d_in[4];
    const float* w_rs  = (const float*)d_in[5];
    const float* bn_rs = (const float*)d_in[6];
    const float* w_off = (const float*)d_in[7];
    const float* b_off = (const float*)d_in[8];
    const float* w_dcn = (const float*)d_in[9];
    const float* b_dcn = (const float*)d_in[10];
    const float* bn1   = (const float*)d_in[11];
    const float* w_c2  = (const float*)d_in[12];
    const float* bn2   = (const float*)d_in[13];
    const float* w_ds  = (const float*)d_in[14];
    const float* bn_ds = (const float*)d_in[15];
    float* out = (float*)d_out;

    const int Cm = 128, H0 = 192, W0 = 192;
    const size_t actB = (size_t)MTOT * Cm * sizeof(float);       // 18.87 MB
    const size_t offB = (size_t)MB * 54 * HWO * sizeof(float);   //  7.96 MB
    const size_t aB   = (size_t)MTOT * 1152 * sizeof(_Float16);  // 84.93 MB

    char* ws = (char*)d_ws;
    size_t cur = 0;
    auto carve = [&](size_t bytes) -> char* {
        char* p = ws + cur;
        cur = (cur + bytes + 255) & ~(size_t)255;
        return p;
    };
    float*    h1   = (float*)carve(actB);   // reused for h3
    float*    t2   = (float*)carve(actB);   // reused for h4
    float*    h2   = (float*)carve(actB);
    float*    offb = (float*)carve(offB);
    _Float16* Abuf = (_Float16*)carve(aB);  // shared im2col / DCN scratch
    _Float16* wh_r1  = (_Float16*)carve(128 * 576  * 2);
    _Float16* wh_r2  = (_Float16*)carve(128 * 1152 * 2);
    _Float16* wh_rs  = (_Float16*)carve(128 * 64   * 2);
    _Float16* wh_off = (_Float16*)carve(54  * 1152 * 2);
    _Float16* wh_dcn = (_Float16*)carve(128 * 1152 * 2);
    _Float16* wh_c2  = (_Float16*)carve(128 * 128  * 2);
    _Float16* wh_ds  = (_Float16*)carve(128 * 64   * 2);
    float* h3 = h1;
    float* h4 = t2;

    auto cvt = [&](const float* s, _Float16* d, int n) {
        wcvt_f16<<<(n + 255) / 256, 256, 0, stream>>>(s, d, n);
    };
    cvt(w_r1,  wh_r1,  128 * 576);
    cvt(w_r2,  wh_r2,  128 * 1152);
    cvt(w_rs,  wh_rs,  128 * 64);
    cvt(w_off, wh_off, 54 * 1152);
    cvt(w_dcn, wh_dcn, 128 * 1152);
    cvt(w_c2,  wh_c2,  128 * 128);
    cvt(w_ds,  wh_ds,  128 * 64);

    const dim3 blk(128);
    const dim3 g128(MTOT / 128, 2);  // 4 waves x 32 rows; Cout=128 -> 2 N tiles
    const dim3 g54 (MTOT / 128, 1);  // Cout=54 -> one guarded N tile
    const dim3 gIm(MTOT);

    // 1) h1 = lrelu(bn_r1(conv3x3 s2 p1 (x)))
    im2col_f16<3, 3><<<gIm, 256, 0, stream>>>(x, Abuf, 64, H0, W0, 2, 1);
    gemm_wmma_f16<576><<<g128, blk, 0, stream>>>(Abuf, wh_r1, bn_r1, nullptr,
                                                 nullptr, h1, 128, 1);
    // 2) t2 = bn_r2(conv3x3 s1 p1 (h1))
    im2col_f16<3, 3><<<gIm, 256, 0, stream>>>(h1, Abuf, 128, WO, WO, 1, 1);
    gemm_wmma_f16<1152><<<g128, blk, 0, stream>>>(Abuf, wh_r2, bn_r2, nullptr,
                                                  nullptr, t2, 128, 0);
    // 3) h2 = lrelu(bn_rs(conv1x1 s2 (x)) + t2)
    im2col_f16<1, 1><<<gIm, 256, 0, stream>>>(x, Abuf, 64, H0, W0, 2, 0);
    gemm_wmma_f16<64><<<g128, blk, 0, stream>>>(Abuf, wh_rs, bn_rs, nullptr,
                                                t2, h2, 128, 1);
    // 4) off = conv3x3 s1 p1 (h2) + b_off
    im2col_f16<3, 3><<<gIm, 256, 0, stream>>>(h2, Abuf, 128, WO, WO, 1, 1);
    gemm_wmma_f16<1152><<<g54, blk, 0, stream>>>(Abuf, wh_off, nullptr, b_off,
                                                 nullptr, offb, 54, 0);
    // 5) modulated bilinear sampling -> f16 im2col (same layout as convs)
    {
        const int total = MB * 2 * 9 * HWO;          // 663552
        mdcn_sample<<<(total + 255) / 256, 256, 0, stream>>>(h2, offb, Abuf);
    }
    // 6) h3 = lrelu(bn1(contract(Abuf, w_dcn) + b_dcn))
    gemm_wmma_f16<1152><<<g128, blk, 0, stream>>>(Abuf, wh_dcn, bn1, b_dcn,
                                                  nullptr, h3, 128, 1);
    // 7) h4 = lrelu(bn2(conv1x1 (h3)))
    im2col_f16<1, 1><<<gIm, 256, 0, stream>>>(h3, Abuf, 128, WO, WO, 1, 0);
    gemm_wmma_f16<128><<<g128, blk, 0, stream>>>(Abuf, wh_c2, bn2, nullptr,
                                                 nullptr, h4, 128, 1);
    // 8) out = bn_ds(conv1x1 s2 (x)) + h4
    im2col_f16<1, 1><<<gIm, 256, 0, stream>>>(x, Abuf, 64, H0, W0, 2, 0);
    gemm_wmma_f16<64><<<g128, blk, 0, stream>>>(Abuf, wh_ds, bn_ds, nullptr,
                                                h4, out, 128, 0);
}
